// BagInput_16621523436170
// MI455X (gfx1250) — matrix operand order repaired and checked by reading, not compiled, not run on hardware
//
#include <hip/hip_runtime.h>
#include <hip/hip_bf16.h>

typedef __attribute__((ext_vector_type(16))) _Float16 v16h;
typedef __attribute__((ext_vector_type(8)))  float    v8f;

#define FEAT 256
#define BAGN 128
#define EPS  1e-5

// ---------------------------------------------------------------------------
// ws layout: [0, 131072)   W fragments (hi/lo f16, WMMA B lane layout)
//            [131072, +1024)  S1[128] double
//            [132096, +1024)  S2[128] double
// d_out doubles as the bag-sum accumulator (zeroed each launch).
// ---------------------------------------------------------------------------

// Zero d_out (bag sums) and the 256 stat doubles.
__global__ void zero_k(float* __restrict__ out, long n, double* __restrict__ stats) {
  long g = (long)blockIdx.x * blockDim.x + threadIdx.x;
  if (g < 256) stats[g] = 0.0;
  long stride = (long)gridDim.x * blockDim.x;
  for (long i = g; i < n; i += stride) out[i] = 0.0f;
}

// Pre-swizzle W[128,256] f32 into hi/lo f16 B-fragments.
// B (32x16, KxN) k order used here (contiguous, same as A): N = lane%16, K = (lane>=16?16:0)+i.
__global__ void wprep_k(const float* __restrict__ W, v16h* __restrict__ bfrag) {
  int t = blockIdx.x * blockDim.x + threadIdx.x;
  if (t >= 2048) return;                     // 64 frag slots * 32 lanes
  int lane = t & 31, slot = t >> 5;
  int nt = slot & 7, kt = slot >> 3;
  int n  = nt * 16 + (lane & 15);
  int kb = kt * 32 + ((lane >> 4) << 4);     // +16 for lanes 16..31
  const float* wr = W + (long)n * FEAT + kb; // W^T[k][n] == W[n][k]
  v16h hi, lo;
#pragma unroll
  for (int i = 0; i < 16; ++i) {
    float wv = wr[i];
    _Float16 h = (_Float16)wv;
    hi[i] = h;
    lo[i] = (_Float16)(wv - (float)h);
  }
  bfrag[(slot * 2 + 0) * 32 + lane] = hi;
  bfrag[(slot * 2 + 1) * 32 + lane] = lo;
}

// Fused: 128x128 GEMM tile via split-f16 WMMA + bias + ReLU + in-register
// run-length segmented sum over the sorted seg_ids (atomics into L2).
__global__ void __launch_bounds__(256)
fused_gemm_seg_k(const float* __restrict__ x, const int* __restrict__ seg,
                 const float* __restrict__ bias, const float* __restrict__ bfragG,
                 float* __restrict__ sums, int n_items) {
  extern __shared__ char smem[];
  v16h*  Bfrag = (v16h*) smem;                     // 131072 B
  int*   segL  = (int*)  (smem + 131072);          //    512 B
  float* biasL = (float*)(smem + 131072 + 512);    //    512 B

  const int tid  = threadIdx.x;
  const int wave = tid >> 5;
  const int lane = tid & 31;
  const long item_base = (long)blockIdx.x * 128;

  // Stage W fragments global -> LDS (128 KB, 512 B per thread).
  {
    const float4* src = (const float4*)bfragG;
    float4*       dst = (float4*)smem;
#pragma unroll
    for (int i = 0; i < 32; ++i) dst[tid + i * 256] = src[tid + i * 256];
  }
  if (tid < 128) {
    long gi = item_base + tid;
    if (gi > (long)n_items - 1) gi = n_items - 1;
    segL[tid]  = seg[gi];
    biasL[tid] = bias[tid];
  }
  __syncthreads();

  v8f acc[8] = {};

  // A-fragment addressing (CDNA5 16-bit A 16x32):
  // M = lane%16; K = (lane>=16?16:0)+i (matches B fragment order)
  const int mrow = wave * 16 + (lane & 15);
  long r = item_base + mrow;
  if (r > (long)n_items - 1) r = n_items - 1;
  const float* xrow = x + r * (long)FEAT;
  const int koff = (lane >> 4) * 16;

#pragma unroll
  for (int kt = 0; kt < 8; ++kt) {
    const float* p0 = xrow + kt * 32 + koff;
    float4 a0 = ((const float4*)p0)[0];
    float4 a1 = ((const float4*)p0)[1];
    float4 a2 = ((const float4*)(p0 + 8))[0];
    float4 a3 = ((const float4*)(p0 + 8))[1];
    float av[16] = {a0.x, a0.y, a0.z, a0.w, a1.x, a1.y, a1.z, a1.w,
                    a2.x, a2.y, a2.z, a2.w, a3.x, a3.y, a3.z, a3.w};
    v16h ah, al;
#pragma unroll
    for (int i = 0; i < 16; ++i) {
      _Float16 h = (_Float16)av[i];
      ah[i] = h;
      al[i] = (_Float16)(av[i] - (float)h);
    }
#pragma unroll
    for (int nt = 0; nt < 8; ++nt) {
      int s = kt * 8 + nt;
      v16h bh = Bfrag[(s * 2 + 0) * 32 + lane];
      v16h bl = Bfrag[(s * 2 + 1) * 32 + lane];
      // hi*hi + lo*hi + hi*lo  (drop lo*lo, ~2^-22 relative)
      acc[nt] = __builtin_amdgcn_wmma_f32_16x16x32_f16(false, ah, false, bh,
                                                       (short)0, acc[nt], false, false);
      acc[nt] = __builtin_amdgcn_wmma_f32_16x16x32_f16(false, al, false, bh,
                                                       (short)0, acc[nt], false, false);
      acc[nt] = __builtin_amdgcn_wmma_f32_16x16x32_f16(false, ah, false, bl,
                                                       (short)0, acc[nt], false, false);
    }
  }

  // C/D layout: element j -> m = j + (lane>=16?8:0), n = lane%16.
  // A lane's 8 accumulator rows are 8 CONSECUTIVE sorted items: do the
  // segmented reduction in registers, one atomic per (bag-run, column).
  const int nloc   = lane & 15;
  const int base_m = wave * 16 + (lane >> 4) * 8;  // row of acc element j=0
#pragma unroll
  for (int nt = 0; nt < 8; ++nt) {
    int n = nt * 16 + nloc;
    float bn = biasL[n];
    int cur = -1;
    float s = 0.0f;
#pragma unroll
    for (int j = 0; j < 8; ++j) {
      long gi = item_base + base_m + j;
      if (gi >= (long)n_items) break;
      int bag = segL[base_m + j];
      float v = acc[nt][j] + bn;
      v = v > 0.0f ? v : 0.0f;
      if (bag != cur) {
        if (cur >= 0)
          __hip_atomic_fetch_add(&sums[(long)cur * BAGN + n], s,
                                 __ATOMIC_RELAXED, __HIP_MEMORY_SCOPE_AGENT);
        cur = bag;
        s = 0.0f;
      }
      s += v;
    }
    if (cur >= 0)
      __hip_atomic_fetch_add(&sums[(long)cur * BAGN + n], s,
                             __ATOMIC_RELAXED, __HIP_MEMORY_SCOPE_AGENT);
  }
}

// Per-feature sum / sumsq of agg = sums/cnt over all bags (double accum).
__global__ void stats_k(const float* __restrict__ sums, const int* __restrict__ cnt,
                        double* __restrict__ S1, double* __restrict__ S2, int n_bags) {
  __shared__ double red[256];
  int tid = threadIdx.x;
  int n = tid & 127, sub = tid >> 7;
  int bag0 = blockIdx.x * 256;
  int bagE = bag0 + 256; if (bagE > n_bags) bagE = n_bags;
  double s1 = 0.0, s2 = 0.0;
  for (int bag = bag0 + sub; bag < bagE; bag += 2) {
    int c = cnt[bag]; if (c < 1) c = 1;
    float a = sums[(long)bag * BAGN + n] / (float)c;
    s1 += (double)a;
    s2 += (double)a * (double)a;
  }
  red[tid] = s1;
  __syncthreads();
  if (sub == 0) {
    double t1 = red[tid] + red[tid + 128];
    __hip_atomic_fetch_add(&S1[n], t1, __ATOMIC_RELAXED, __HIP_MEMORY_SCOPE_AGENT);
  }
  __syncthreads();
  red[tid] = s2;
  __syncthreads();
  if (sub == 0) {
    double t2 = red[tid] + red[tid + 128];
    __hip_atomic_fetch_add(&S2[n], t2, __ATOMIC_RELAXED, __HIP_MEMORY_SCOPE_AGENT);
  }
}

// Finalize in place: out = (sums/cnt - mean) * rsqrt(var+eps) * gamma + beta.
__global__ void bn_k(float* __restrict__ out, const int* __restrict__ cnt,
                     const float* __restrict__ gamma, const float* __restrict__ beta,
                     const double* __restrict__ S1, const double* __restrict__ S2,
                     long total, int n_bags) {
  long g = (long)blockIdx.x * blockDim.x + threadIdx.x;
  if (g >= total) return;
  int  n   = (int)(g & 127);
  long bag = g >> 7;
  double mean = S1[n] / (double)n_bags;
  double var  = S2[n] / (double)n_bags - mean * mean;
  if (var < 0.0) var = 0.0;
  double inv = 1.0 / sqrt(var + (double)EPS);
  int c = cnt[bag]; if (c < 1) c = 1;
  float a = out[g] / (float)c;
  out[g] = (float)(((double)a - mean) * inv) * gamma[n] + beta[n];
}

extern "C" void kernel_launch(void* const* d_in, const int* in_sizes, int n_in,
                              void* d_out, int out_size, void* d_ws, size_t ws_size,
                              hipStream_t stream) {
  const float* x     = (const float*)d_in[0];
  const float* W     = (const float*)d_in[1];
  const float* b     = (const float*)d_in[2];
  const float* gamma = (const float*)d_in[3];
  const float* beta  = (const float*)d_in[4];
  const int*   seg   = (const int*)d_in[5];
  const int*   cnt   = (const int*)d_in[6];

  int n_items = in_sizes[0] / FEAT;
  int n_bags  = in_sizes[6];
  float* out  = (float*)d_out;

  char*   ws    = (char*)d_ws;
  v16h*   bfrag = (v16h*)ws;
  double* S1    = (double*)(ws + 131072);
  double* S2    = S1 + 128;

  long total = (long)n_bags * BAGN;

  zero_k<<<2048, 256, 0, stream>>>(out, total, S1);
  wprep_k<<<8, 256, 0, stream>>>(W, bfrag);

  int nblk = (n_items + 127) / 128;
  size_t smem = 131072 + 512 + 512;   // 132096 B: fits even a 160 KB CU-mode cap
  fused_gemm_seg_k<<<nblk, 256, smem, stream>>>(x, seg, b, (const float*)bfrag,
                                                out, n_items);

  stats_k<<<(n_bags + 255) / 256, 256, 0, stream>>>(out, cnt, S1, S2, n_bags);
  bn_k<<<(int)((total + 255) / 256), 256, 0, stream>>>(out, cnt, gamma, beta,
                                                       S1, S2, total, n_bags);
}
